// our_model_39762807226729
// MI455X (gfx1250) — matrix operand order repaired and checked by reading, not compiled, not run on hardware
//
#include <hip/hip_runtime.h>

// ---------------------------------------------------------------------------
// Cross-modal attention block for MI455X (gfx1250).
// fp32 WMMA (16x16x4) for projection/FC GEMMs; bf16 WMMA (16x16x32, f32
// accumulate) for the dominant TxT score/mix GEMMs. bf16 fragments load
// straight from row-major (A) / n-major (B^T) memory, 16 contiguous k per lane half (same order for A and B).
// The mix GEMMs stage their block-shared B tiles in LDS via the gfx1250
// async load-to-LDS path (double buffered, ASYNCcnt), with a synchronous
// LDS fallback if the builtins are unavailable.
// Workspace: ~202 MB.
// ---------------------------------------------------------------------------

typedef __attribute__((ext_vector_type(2)))  float v2f;
typedef __attribute__((ext_vector_type(8)))  float v8f;
typedef __bf16 bf16_t;
typedef __attribute__((ext_vector_type(8)))  bf16_t v8bf;
typedef __attribute__((ext_vector_type(16))) bf16_t v16bf;
typedef int v4i __attribute__((__vector_size__(16)));

#define BB 8
#define TT 2048
#define DD 256
#define BT (BB * TT)
#define EPS_SUM 1e-8f
#define EPS_LN  1e-6f

#define AS3 __attribute__((address_space(3)))

#if __has_builtin(__builtin_amdgcn_global_load_async_to_lds_b128) && \
    __has_builtin(__builtin_amdgcn_s_wait_asynccnt)
#define HAS_ASYNC 1
#else
#define HAS_ASYNC 0
#endif

// byte offsets into workspace
constexpr size_t SZ_PROJ  = (size_t)BT * DD * 2;            // 8,388,608
constexpr size_t SZ_S     = (size_t)BB * TT * TT * 2;       // 67,108,864
constexpr size_t OFF_VB1T = 0;                              // bf16 [b][h][t]
constexpr size_t OFF_VB2  = SZ_PROJ;                        // bf16 [b][t][h]
constexpr size_t OFF_AB1  = 2 * SZ_PROJ;                    // bf16 [b][t][h]
constexpr size_t OFF_AB2T = 3 * SZ_PROJ;                    // bf16 [b][h][t]
constexpr size_t OFF_S    = 4 * SZ_PROJ;                    // bf16 [b][v][a]; gammaA in place
constexpr size_t OFF_GB   = OFF_S + SZ_S;                   // bf16 [b][a][v]
constexpr size_t OFF_CUTA = OFF_GB + SZ_S;                  // f32 [BT]
constexpr size_t OFF_RDA  = OFF_CUTA + (size_t)BT * 4;
constexpr size_t OFF_CUTB = OFF_RDA + (size_t)BT * 4;
constexpr size_t OFF_RDB  = OFF_CUTB + (size_t)BT * 4;
constexpr size_t OFF_VPRE = OFF_RDB + (size_t)BT * 4;       // f32 [BT][D]
constexpr size_t OFF_APRE = OFF_VPRE + (size_t)BT * DD * 4;
constexpr size_t OFF_FCV  = OFF_GB;                         // reuse after mix
constexpr size_t OFF_FCA  = OFF_GB + (size_t)BT * DD * 4;
// total = OFF_APRE + BT*DD*4 = 201,588,736 bytes

__device__ __forceinline__ v8f wmma_f32(v2f a, v2f b, v8f c) {
  return __builtin_amdgcn_wmma_f32_16x16x4_f32(false, a, false, b, (short)0, c,
                                               false, false);
}
__device__ __forceinline__ v8f wmma_bf16(v16bf a, v16bf b, v8f c) {
  return __builtin_amdgcn_wmma_f32_16x16x32_bf16(false, a, false, b, (short)0, c,
                                                 false, false);
}

// A fragment (16x32 bf16) from row-major tile (row stride ld elements)
__device__ __forceinline__ v16bf ld_afrag(const bf16_t* tile, size_t ld, int lane, int k0) {
  const bf16_t* p = tile + (size_t)(lane & 15) * ld + k0 + ((lane >> 4) << 4);
  v8bf lo = *reinterpret_cast<const v8bf*>(p);
  v8bf hi = *reinterpret_cast<const v8bf*>(p + 8);
  return __builtin_shufflevector(lo, hi, 0, 1, 2, 3, 4, 5, 6, 7,
                                 8, 9, 10, 11, 12, 13, 14, 15);
}
// B fragment (32x16 bf16) from B^T stored row-major (n-major, row stride ld)
__device__ __forceinline__ v16bf ld_bfrag(const bf16_t* tileT, size_t ld, int lane, int k0) {
  const bf16_t* p = tileT + (size_t)(lane & 15) * ld + k0 + ((lane >> 4) << 4);
  v8bf lo = *reinterpret_cast<const v8bf*>(p);
  v8bf hi = *reinterpret_cast<const v8bf*>(p + 8);
  return __builtin_shufflevector(lo, hi, 0, 1, 2, 3, 4, 5, 6, 7,
                                 8, 9, 10, 11, 12, 13, 14, 15);
}
// B fragment from an LDS-staged tile (16 contiguous k per lane)
__device__ __forceinline__ v16bf ld_bfrag_lds(const bf16_t* p) {
  v8bf lo = *reinterpret_cast<const v8bf*>(p);
  v8bf hi = *reinterpret_cast<const v8bf*>(p + 8);
  return __builtin_shufflevector(lo, hi, 0, 1, 2, 3, 4, 5, 6, 7,
                                 8, 9, 10, 11, 12, 13, 14, 15);
}

// stage 32 bytes (16 bf16) global -> LDS, async if available
__device__ __forceinline__ void stage32(const bf16_t* gsrc, bf16_t* ldst) {
#if HAS_ASYNC
  __builtin_amdgcn_global_load_async_to_lds_b128((v4i*)gsrc, (AS3 v4i*)ldst, 0, 0);
  __builtin_amdgcn_global_load_async_to_lds_b128((v4i*)gsrc, (AS3 v4i*)ldst, 16, 0);
#else
  v8bf x0 = *reinterpret_cast<const v8bf*>(gsrc);
  v8bf x1 = *reinterpret_cast<const v8bf*>(gsrc + 8);
  *reinterpret_cast<v8bf*>(ldst) = x0;
  *reinterpret_cast<v8bf*>(ldst + 8) = x1;
#endif
}
__device__ __forceinline__ void wait_stage() {
#if HAS_ASYNC
  __builtin_amdgcn_s_wait_asynccnt(0);
#endif
}

// ---- 1) projections (fp32 WMMA), outputs bf16 (two of them transposed) -----
__global__ __launch_bounds__(128) void proj_kernel(
    const float* __restrict__ vf, const float* __restrict__ af,
    const float* __restrict__ Wv1, const float* __restrict__ Wv2,
    const float* __restrict__ Wa1, const float* __restrict__ Wa2,
    char* __restrict__ ws) {
  const int wave = threadIdx.x >> 5, lane = threadIdx.x & 31;
  const int m0 = (blockIdx.x * 4 + wave) * 16;
  const int n0 = blockIdx.y * 16;
  const int which = blockIdx.z;  // 0:vb1T 1:vb2 2:ab1 3:ab2T
  const float* X = (which < 2) ? vf : af;
  const float* W = (which == 0) ? Wv1 : (which == 1) ? Wv2 : (which == 2) ? Wa1 : Wa2;
  const int m = lane & 15, kp = (lane >> 4) << 1;
  const float* Arow = X + (size_t)(m0 + m) * DD;
  const float* Brow = W + (size_t)(n0 + m) * DD;
  v8f acc = {};
  for (int k = 0; k < DD; k += 4) {
    v2f a2 = *reinterpret_cast<const v2f*>(Arow + k + kp);
    v2f b2 = *reinterpret_cast<const v2f*>(Brow + k + kp);
    acc = wmma_f32(a2, b2, acc);
  }
  const int rbase = (lane >> 4) * 8, col = lane & 15;
  if (which == 1 || which == 2) {           // natural [t][h] bf16
    bf16_t* out = (bf16_t*)(ws + (which == 1 ? OFF_VB2 : OFF_AB1));
#pragma unroll
    for (int r = 0; r < 8; ++r) {
      float v = acc[r];
      out[(size_t)(m0 + rbase + r) * DD + n0 + col] = (bf16_t)(v > 0.f ? v : 0.f);
    }
  } else {                                   // transposed [b][h][t] bf16
    bf16_t* out = (bf16_t*)(ws + (which == 0 ? OFF_VB1T : OFF_AB2T));
    const int b = m0 / TT;
    const int t0 = (m0 % TT) + rbase;
    v8bf pk;
#pragma unroll
    for (int r = 0; r < 8; ++r) {
      float v = acc[r];
      pk[r] = (bf16_t)(v > 0.f ? v : 0.f);
    }
    *reinterpret_cast<v8bf*>(out + ((size_t)b * DD + n0 + col) * TT + t0) = pk;
  }
}

// ---- 2) scores (bf16 WMMA): S = relu(vb2 . ab1 / 16), bf16 ------------------
__global__ __launch_bounds__(128) void scores_kernel(char* __restrict__ ws) {
  const bf16_t* vb2 = (const bf16_t*)(ws + OFF_VB2);
  const bf16_t* ab1 = (const bf16_t*)(ws + OFF_AB1);
  bf16_t* S = (bf16_t*)(ws + OFF_S);
  const int wave = threadIdx.x >> 5, lane = threadIdx.x & 31;
  const int b = blockIdx.z;
  const int v0 = (blockIdx.x * 4 + wave) * 16;
  const int a0 = blockIdx.y * 64;            // 4 a-tiles per wave (A reuse)
  const bf16_t* Ab = vb2 + ((size_t)b * TT + v0) * DD;
  const bf16_t* Bb = ab1 + ((size_t)b * TT + a0) * DD;
  v8f c0 = {}, c1 = {}, c2 = {}, c3 = {};
#pragma unroll
  for (int k = 0; k < DD; k += 32) {
    v16bf a  = ld_afrag(Ab, DD, lane, k);
    v16bf b0 = ld_bfrag(Bb,           DD, lane, k);
    v16bf b1 = ld_bfrag(Bb + 16 * DD, DD, lane, k);
    v16bf b2 = ld_bfrag(Bb + 32 * DD, DD, lane, k);
    v16bf b3 = ld_bfrag(Bb + 48 * DD, DD, lane, k);
    c0 = wmma_bf16(a, b0, c0);
    c1 = wmma_bf16(a, b1, c1);
    c2 = wmma_bf16(a, b2, c2);
    c3 = wmma_bf16(a, b3, c3);
  }
  const int rbase = (lane >> 4) * 8, col = lane & 15;
#pragma unroll
  for (int r = 0; r < 8; ++r) {
    size_t o = ((size_t)b * TT + v0 + rbase + r) * TT + a0 + col;
    float x0 = c0[r] * 0.0625f, x1 = c1[r] * 0.0625f;
    float x2 = c2[r] * 0.0625f, x3 = c3[r] * 0.0625f;
    S[o]      = (bf16_t)(x0 > 0.f ? x0 : 0.f);
    S[o + 16] = (bf16_t)(x1 > 0.f ? x1 : 0.f);
    S[o + 32] = (bf16_t)(x2 > 0.f ? x2 : 0.f);
    S[o + 48] = (bf16_t)(x3 > 0.f ? x3 : 0.f);
  }
}

// ---- 3a) row stats: cutA/rdA per (b,v) --------------------------------------
__global__ __launch_bounds__(256) void row_stats_kernel(
    const float* __restrict__ thr_p, char* __restrict__ ws) {
  const int v = blockIdx.x, b = blockIdx.y, t = threadIdx.x;
  const bf16_t* S = (const bf16_t*)(ws + OFF_S);
  const v8bf* srow = (const v8bf*)(S + ((size_t)b * TT + v) * TT);
  v8bf x8 = srow[t];                          // 256 threads * 8 = 2048
  float xs[8];
  float acc = 0.f;
#pragma unroll
  for (int j = 0; j < 8; ++j) { xs[j] = (float)x8[j]; acc += xs[j]; }
  __shared__ float red[256];
  red[t] = acc; __syncthreads();
  for (int o = 128; o > 0; o >>= 1) { if (t < o) red[t] += red[t + o]; __syncthreads(); }
  const float rs = red[0] + EPS_SUM;
  const float cut = thr_p[0] * rs;
  __syncthreads();
  float acc2 = 0.f;
#pragma unroll
  for (int j = 0; j < 8; ++j) acc2 += (xs[j] > cut) ? xs[j] : 0.f;
  red[t] = acc2; __syncthreads();
  for (int o = 128; o > 0; o >>= 1) { if (t < o) red[t] += red[t + o]; __syncthreads(); }
  if (t == 0) {
    ((float*)(ws + OFF_CUTA))[(size_t)b * TT + v] = cut;
    ((float*)(ws + OFF_RDA))[(size_t)b * TT + v] = 1.f / (red[0] + EPS_SUM * rs);
  }
}

// ---- 3b) col stats: cutB/rdB per (b,a); 64 cols x 4 v-chunks per block ------
__global__ __launch_bounds__(256) void col_stats_kernel(
    const float* __restrict__ thr_p, char* __restrict__ ws) {
  const bf16_t* S = (const bf16_t*)(ws + OFF_S);
  const int b = blockIdx.y;
  const int c = threadIdx.x & 63;
  const int chunk = threadIdx.x >> 6;         // 0..3, 512 rows each
  const int a = blockIdx.x * 64 + c;
  const bf16_t* Sb = S + (size_t)b * TT * TT + a;
  __shared__ float red[256];
  __shared__ float cuts[64];
  __shared__ float css[64];
  float acc = 0.f;
  for (int v = chunk * 512; v < chunk * 512 + 512; ++v) acc += (float)Sb[(size_t)v * TT];
  red[threadIdx.x] = acc; __syncthreads();
  if (threadIdx.x < 64) {
    float cs = red[c] + red[c + 64] + red[c + 128] + red[c + 192] + EPS_SUM;
    css[c] = cs;
    cuts[c] = thr_p[0] * cs;
  }
  __syncthreads();
  const float cut = cuts[c];
  float acc2 = 0.f;
  for (int v = chunk * 512; v < chunk * 512 + 512; ++v) {
    float x = (float)Sb[(size_t)v * TT];
    acc2 += (x > cut) ? x : 0.f;
  }
  __syncthreads();
  red[threadIdx.x] = acc2; __syncthreads();
  if (threadIdx.x < 64) {
    float ts = red[c] + red[c + 64] + red[c + 128] + red[c + 192];
    ((float*)(ws + OFF_CUTB))[(size_t)b * TT + a] = cut;
    ((float*)(ws + OFF_RDB))[(size_t)b * TT + a] = 1.f / (ts + EPS_SUM * css[c]);
  }
}

// ---- 4) gammaB = transform(S^T) via 16x16 LDS tile transpose ---------------
__global__ __launch_bounds__(256) void gammaB_kernel(char* __restrict__ ws) {
  __shared__ bf16_t tile[16][17];
  const bf16_t* S = (const bf16_t*)(ws + OFF_S);
  bf16_t* gB = (bf16_t*)(ws + OFF_GB);
  const int b = blockIdx.z, v0 = blockIdx.y * 16, a0 = blockIdx.x * 16;
  const int r = threadIdx.x >> 4, c = threadIdx.x & 15;
  tile[r][c] = S[((size_t)b * TT + v0 + r) * TT + a0 + c];
  __syncthreads();
  const float cut = ((const float*)(ws + OFF_CUTB))[(size_t)b * TT + a0 + r];
  const float rd  = ((const float*)(ws + OFF_RDB))[(size_t)b * TT + a0 + r];
  const float f = (float)tile[c][r];
  gB[((size_t)b * TT + a0 + r) * TT + v0 + c] = (bf16_t)(f > cut ? f * rd : 0.f);
}

// ---- 5) gammaA: in-place transform of S -------------------------------------
__global__ __launch_bounds__(256) void gammaA_kernel(char* __restrict__ ws) {
  const size_t idx = (size_t)blockIdx.x * 256 + threadIdx.x;  // v8bf index
  const int row = (int)(idx >> 8);                            // 256 vecs per row
  v8bf* S8 = (v8bf*)(ws + OFF_S);
  v8bf x = S8[idx];
  const float cut = ((const float*)(ws + OFF_CUTA))[row];
  const float rd  = ((const float*)(ws + OFF_RDA))[row];
#pragma unroll
  for (int j = 0; j < 8; ++j) {
    float f = (float)x[j];
    x[j] = (bf16_t)(f > cut ? f * rd : 0.f);
  }
  S8[idx] = x;
}

// ---- 6+7) mix: out = resid + gamma @ B  (bf16 WMMA, B staged in LDS) --------
// z = blockIdx.z: b = z&7, sel = z>>3 (0: vpre from gammaA/ab2T, 1: apre from gammaB/vb1T)
__global__ __launch_bounds__(128) void mix_kernel(
    const float* __restrict__ vf, const float* __restrict__ af,
    char* __restrict__ ws) {
  const int z = blockIdx.z;
  const int b = z & (BB - 1);
  const int sel = z >> 3;
  const bf16_t* Amat = (const bf16_t*)(ws + (sel == 0 ? OFF_S : OFF_GB));
  const bf16_t* Bmat = (const bf16_t*)(ws + (sel == 0 ? OFF_AB2T : OFF_VB1T));
  const float* resid = sel == 0 ? vf : af;
  float* outp = (float*)(ws + (sel == 0 ? OFF_VPRE : OFF_APRE));

  const int wave = threadIdx.x >> 5, lane = threadIdx.x & 31;
  const int m0 = (blockIdx.x * 4 + wave) * 16;   // row tile (v or a)
  const int h0 = blockIdx.y * 64;                // 4 feature tiles
  const bf16_t* Ab = Amat + ((size_t)b * TT + m0) * TT;
  const bf16_t* Bt = Bmat + ((size_t)b * DD + h0) * TT;

  // LDS double buffer: 64 B^T rows x 64 k, stride 72 (conflict-free frags)
  __shared__ bf16_t Bs[2][64][72];
  const int srow = threadIdx.x >> 1;             // 0..63
  const int shalf = (threadIdx.x & 1) * 16;      // 0 or 16 elements
  const bf16_t* gstage = Bt + (size_t)srow * TT + shalf;
  const bf16_t* Arow = Ab + (size_t)(lane & 15) * TT;  // prefetch stream

  v8f c0 = {}, c1 = {}, c2 = {}, c3 = {};
  stage32(gstage, &Bs[0][srow][shalf]);
  for (int kc = 0; kc < TT; kc += 64) {
    const int buf = (kc >> 6) & 1;
    wait_stage();            // my staged data for `buf` has landed
    __syncthreads();         // everyone's landed; everyone done reading buf^1
    if (kc + 64 < TT) stage32(gstage + kc + 64, &Bs[buf ^ 1][srow][shalf]);
    __builtin_prefetch(Arow + kc + 128, 0, 3);
#pragma unroll
    for (int ki = 0; ki < 2; ++ki) {
      const int k = kc + ki * 32;
      v16bf a = ld_afrag(Ab, TT, lane, k);
      const int kl = ki * 32 + ((lane >> 4) << 4);
      const int n = lane & 15;
      v16bf b0 = ld_bfrag_lds(&Bs[buf][n][kl]);
      v16bf b1 = ld_bfrag_lds(&Bs[buf][16 + n][kl]);
      v16bf b2 = ld_bfrag_lds(&Bs[buf][32 + n][kl]);
      v16bf b3 = ld_bfrag_lds(&Bs[buf][48 + n][kl]);
      c0 = wmma_bf16(a, b0, c0);
      c1 = wmma_bf16(a, b1, c1);
      c2 = wmma_bf16(a, b2, c2);
      c3 = wmma_bf16(a, b3, c3);
    }
  }
  const int rbase = (lane >> 4) * 8, col = lane & 15;
#pragma unroll
  for (int r = 0; r < 8; ++r) {
    size_t o = ((size_t)b * TT + m0 + rbase + r) * DD + h0 + col;
    outp[o]      = c0[r] + resid[o];
    outp[o + 16] = c1[r] + resid[o + 16];
    outp[o + 32] = c2[r] + resid[o + 32];
    outp[o + 48] = c3[r] + resid[o + 48];
  }
}

// ---- 8) final FC (fp32 WMMA): fc = relu(pre @ Wfc^T) ------------------------
__global__ __launch_bounds__(128) void fc_kernel(
    const float* __restrict__ Wvfc, const float* __restrict__ Wac,
    char* __restrict__ ws) {
  const int wave = threadIdx.x >> 5, lane = threadIdx.x & 31;
  const int m0 = (blockIdx.x * 4 + wave) * 16;
  const int n0 = blockIdx.y * 16;
  const int zz = blockIdx.z;
  const float* X = (const float*)(ws + (zz == 0 ? OFF_VPRE : OFF_APRE));
  const float* W = zz == 0 ? Wvfc : Wac;
  float* out = (float*)(ws + (zz == 0 ? OFF_FCV : OFF_FCA));
  const int m = lane & 15, kp = (lane >> 4) << 1;
  const float* Arow = X + (size_t)(m0 + m) * DD;
  const float* Brow = W + (size_t)(n0 + m) * DD;
  v8f acc = {};
  for (int k = 0; k < DD; k += 4) {
    v2f a2 = *reinterpret_cast<const v2f*>(Arow + k + kp);
    v2f b2 = *reinterpret_cast<const v2f*>(Brow + k + kp);
    acc = wmma_f32(a2, b2, acc);
  }
  const int rbase = (lane >> 4) * 8, col = lane & 15;
#pragma unroll
  for (int r = 0; r < 8; ++r) {
    float v = acc[r];
    out[(size_t)(m0 + rbase + r) * DD + n0 + col] = v > 0.f ? v : 0.f;
  }
}

// ---- 9) layernorm both + combine, write all three outputs -------------------
__global__ __launch_bounds__(256) void ln_kernel(
    const float* __restrict__ lns, const float* __restrict__ lnb,
    const char* __restrict__ ws, float* __restrict__ out) {
  const int r = blockIdx.x, t = threadIdx.x;
  const float xv = ((const float*)(ws + OFF_FCV))[(size_t)r * DD + t];
  const float xa = ((const float*)(ws + OFF_FCA))[(size_t)r * DD + t];
  __shared__ float s0[256], s1[256], s2[256], s3[256];
  s0[t] = xv; s1[t] = xv * xv; s2[t] = xa; s3[t] = xa * xa;
  __syncthreads();
  for (int o = 128; o > 0; o >>= 1) {
    if (t < o) { s0[t] += s0[t + o]; s1[t] += s1[t + o]; s2[t] += s2[t + o]; s3[t] += s3[t + o]; }
    __syncthreads();
  }
  const float inv = 1.0f / DD;
  const float muv = s0[0] * inv, mua = s2[0] * inv;
  const float varv = s1[0] * inv - muv * muv;
  const float vara = s3[0] * inv - mua * mua;
  const float rv = rsqrtf(varv + EPS_LN), ra = rsqrtf(vara + EPS_LN);
  const float g = lns[t], bia = lnb[t];
  const float vo = (xv - muv) * rv * g + bia;
  const float ao = (xa - mua) * ra * g + bia;
  const size_t o = (size_t)r * DD + t;
  out[o] = 0.4f * (vo + ao);                    // av_f
  out[(size_t)BT * DD + o] = vo;                // v_out
  out[2 * (size_t)BT * DD + o] = ao;            // a_out
}

extern "C" void kernel_launch(void* const* d_in, const int* in_sizes, int n_in,
                              void* d_out, int out_size, void* d_ws, size_t ws_size,
                              hipStream_t stream) {
  (void)in_sizes; (void)n_in; (void)out_size; (void)ws_size;
  const float* a_f  = (const float*)d_in[0];
  const float* v_f  = (const float*)d_in[1];
  const float* thr  = (const float*)d_in[2];
  const float* Wv1  = (const float*)d_in[3];
  const float* Wv2  = (const float*)d_in[4];
  const float* Wvfc = (const float*)d_in[5];
  const float* Wa1  = (const float*)d_in[6];
  const float* Wa2  = (const float*)d_in[7];
  const float* Wac  = (const float*)d_in[8];
  const float* lns  = (const float*)d_in[9];
  const float* lnb  = (const float*)d_in[10];
  char* ws = (char*)d_ws;
  float* out = (float*)d_out;

  proj_kernel<<<dim3(BT / 16 / 4, DD / 16, 4), 128, 0, stream>>>(v_f, a_f, Wv1, Wv2, Wa1, Wa2, ws);
  scores_kernel<<<dim3(TT / 16 / 4, TT / 64, BB), 128, 0, stream>>>(ws);
  row_stats_kernel<<<dim3(TT, BB), 256, 0, stream>>>(thr, ws);
  col_stats_kernel<<<dim3(TT / 64, BB), 256, 0, stream>>>(thr, ws);
  gammaB_kernel<<<dim3(TT / 16, TT / 16, BB), 256, 0, stream>>>(ws);   // reads S pre-transform
  gammaA_kernel<<<dim3((unsigned)((size_t)BB * TT * TT / 8 / 256)), 256, 0, stream>>>(ws);
  mix_kernel<<<dim3(TT / 16 / 4, DD / 64, BB * 2), 128, 0, stream>>>(v_f, a_f, ws);
  fc_kernel<<<dim3(BT / 16 / 4, DD / 16, 2), 128, 0, stream>>>(Wvfc, Wac, ws);
  ln_kernel<<<dim3(BT), 256, 0, stream>>>(lns, lnb, ws, out);
}
